// MultiHeadSelfAttention_11562051961171
// MI455X (gfx1250) — compile-verified, matrix operand order checked by reading, not run on hardware
//
#include <hip/hip_runtime.h>
#include <hip/hip_bf16.h>

// ---- types ---------------------------------------------------------------
typedef __attribute__((ext_vector_type(4)))  __bf16 v4bf;
typedef __attribute__((ext_vector_type(8)))  __bf16 v8bf;
typedef __attribute__((ext_vector_type(16))) __bf16 v16bf;
typedef __attribute__((ext_vector_type(8)))  float  v8f;

// B=4, L=2048, D=1024, H=16, Hd=64

// ---- WMMA helper ---------------------------------------------------------
__device__ __forceinline__ v8f wmma_bf16(v16bf a, v16bf b, v8f c) {
    return __builtin_amdgcn_wmma_f32_16x16x32_bf16(
        false, a, false, b, (short)0, c, false, false);
}

// 16x32 bf16 fragment; each "row" contiguous with given element stride.
//   lanes 0-15 : row = lane,     cols 0-7 / 16-23
//   lanes 16-31: row = lane-16,  cols 8-15 / 24-31
__device__ __forceinline__ v16bf load_frag(const __bf16* __restrict__ tile,
                                           int stride, int lane) {
    const int hi = lane >> 4;
    const int r  = lane & 15;
    const __bf16* p = tile + (long)r * stride + hi * 8;
    v8bf c0 = *(const v8bf*)(p);
    v8bf c1 = *(const v8bf*)(p + 16);
    v16bf a;
#pragma unroll
    for (int i = 0; i < 8; ++i) { a[i] = c0[i]; a[i + 8] = c1[i]; }
    return a;
}

// CDNA5 async copy: 16 bytes global -> LDS per active lane, ASYNCcnt-tracked.
__device__ __forceinline__ void async_b128(unsigned lds_off,
                                           unsigned long long gaddr) {
    asm volatile("global_load_async_to_lds_b128 %0, %1, off"
                 :: "v"(lds_off), "v"(gaddr) : "memory");
}

// ---- fp32 -> bf16 conversion --------------------------------------------
__global__ __launch_bounds__(256) void k_cvt(const float* __restrict__ in,
                                             __bf16* __restrict__ out, int n) {
    int i = (blockIdx.x * 256 + threadIdx.x) * 4;
    if (i + 3 < n) {
        float4 f = *(const float4*)(in + i);
        v4bf o;
        o[0] = (__bf16)f.x; o[1] = (__bf16)f.y;
        o[2] = (__bf16)f.z; o[3] = (__bf16)f.w;
        *(v4bf*)(out + i) = o;
    }
}

// ---- pipelined 32x64 GEMM core: acc += X(32x1024) @ W(64x1024)^T ---------
// K-loop unrolled by 2 with two fragment register sets ping-ponging so the
// double buffer costs zero register moves; loads for step k+1 issue while
// step k's WMMAs execute (partial s_wait_loadcnt).
__device__ __forceinline__ void gemm_core(const __bf16* __restrict__ x0,
                                          const __bf16* __restrict__ w0,
                                          int lane, v8f acc[2][4]) {
    v16bf a0[2], b0[4], a1[2], b1[4];
    a0[0] = load_frag(x0, 1024, lane);
    a0[1] = load_frag(x0 + 16 * 1024, 1024, lane);
#pragma unroll
    for (int s = 0; s < 4; ++s)
        b0[s] = load_frag(w0 + (long)s * 16 * 1024, 1024, lane);

    for (int kk = 0; kk < 1024; kk += 64) {
        const int k1 = kk + 32;
        a1[0] = load_frag(x0 + k1, 1024, lane);
        a1[1] = load_frag(x0 + 16 * 1024 + k1, 1024, lane);
#pragma unroll
        for (int s = 0; s < 4; ++s)
            b1[s] = load_frag(w0 + (long)s * 16 * 1024 + k1, 1024, lane);
#pragma unroll
        for (int s = 0; s < 4; ++s) {
            acc[0][s] = wmma_bf16(a0[0], b0[s], acc[0][s]);
            acc[1][s] = wmma_bf16(a0[1], b0[s], acc[1][s]);
        }
        const int k2 = (kk + 64) & 1023;       // final prefetch wraps (dummy)
        a0[0] = load_frag(x0 + k2, 1024, lane);
        a0[1] = load_frag(x0 + 16 * 1024 + k2, 1024, lane);
#pragma unroll
        for (int s = 0; s < 4; ++s)
            b0[s] = load_frag(w0 + (long)s * 16 * 1024 + k2, 1024, lane);
#pragma unroll
        for (int s = 0; s < 4; ++s) {
            acc[0][s] = wmma_bf16(a1[0], b1[s], acc[0][s]);
            acc[1][s] = wmma_bf16(a1[1], b1[s], acc[1][s]);
        }
    }
}

// ---- QKV projection ------------------------------------------------------
//   vt_layout==0 : [B,H,L,Hd]   (Q, K)
//   vt_layout==1 : [B,H,Hd,L]   (V transposed, so PV B-frags are contiguous)
__global__ __launch_bounds__(256) void k_gemm_qkv(
    const __bf16* __restrict__ X, const __bf16* __restrict__ W,
    __bf16* __restrict__ out, float scale, int vt_layout) {
    const int lane = threadIdx.x & 31;
    const int wid  = threadIdx.x >> 5;
    const int gw   = blockIdx.x * 8 + wid;   // 4096 waves
    const int tm   = gw >> 4;                // 256 M-tiles (32 rows)
    const int tn   = gw & 15;                // 16  N-tiles (64 cols)

    v8f acc[2][4] = {};
    gemm_core(X + (long)tm * 32 * 1024, W + (long)tn * 64 * 1024, lane, acc);

    const int hi = lane >> 4, nl = lane & 15;
#pragma unroll
    for (int mi = 0; mi < 2; ++mi) {
#pragma unroll
        for (int s = 0; s < 4; ++s) {
#pragma unroll
            for (int r = 0; r < 8; ++r) {
                int m = tm * 32 + mi * 16 + hi * 8 + r;   // b*L + l
                int n = tn * 64 + s * 16 + nl;            // h*Hd + hd
                int b = m >> 11, l = m & 2047;
                int h = n >> 6,  hd = n & 63;
                long idx = vt_layout
                             ? ((long)((b * 16 + h) * 64 + hd) * 2048 + l)
                             : ((long)((b * 16 + h) * 2048 + l) * 64 + hd);
                out[idx] = (__bf16)(acc[mi][s][r] * scale);
            }
        }
    }
}

// ---- output projection: d_out = CTX @ Wo^T (f32 out) ---------------------
__global__ __launch_bounds__(256) void k_gemm_out(
    const __bf16* __restrict__ X, const __bf16* __restrict__ W,
    float* __restrict__ out) {
    const int lane = threadIdx.x & 31;
    const int wid  = threadIdx.x >> 5;
    const int gw   = blockIdx.x * 8 + wid;
    const int tm   = gw >> 4;
    const int tn   = gw & 15;

    v8f acc[2][4] = {};
    gemm_core(X + (long)tm * 32 * 1024, W + (long)tn * 64 * 1024, lane, acc);

    const int hi = lane >> 4, nl = lane & 15;
#pragma unroll
    for (int mi = 0; mi < 2; ++mi) {
#pragma unroll
        for (int s = 0; s < 4; ++s) {
#pragma unroll
            for (int r = 0; r < 8; ++r) {
                long m = tm * 32 + mi * 16 + hi * 8 + r;
                long n = tn * 64 + s * 16 + nl;
                out[m * 1024 + n] = acc[mi][s][r];
            }
        }
    }
}

// ---- T5 relative position bias ------------------------------------------
__device__ __forceinline__ float rel_bias(int rp, int h,
                                          const float* __restrict__ remb) {
    int sb = (rp > 0) ? 16 : 0;
    int d  = rp < 0 ? -rp : rp;
    int bucket;
    if (d < 8) {
        bucket = d;
    } else {
        float lr = __logf((float)d * 0.125f) * 0.36067376022224085f; // /ln16
        int lg = 8 + (int)(lr * 8.0f);
        bucket = lg < 15 ? lg : 15;
    }
    return remb[(bucket + sb) * 16 + h];
}

// ---- flash attention -----------------------------------------------------
// One wave per (b, h, 16-row q tile); the 8 waves of a block share one
// (b, h) and therefore the SAME K/V tiles.  Each 32-key chunk's K (32x64)
// and V^T (64x32) tiles are staged into LDS once per block with
// global_load_async_to_lds_b128 (1 async op per thread per tile),
// double-buffered: chunk kc+1 copies in flight while chunk kc computes.
__global__ __launch_bounds__(256) void k_attn(
    const __bf16* __restrict__ Q, const __bf16* __restrict__ K,
    const __bf16* __restrict__ Vt, const int* __restrict__ mask,
    const float* __restrict__ rel, __bf16* __restrict__ ctx) {
    __shared__ __align__(16) __bf16 ksm[2][32 * 64];   // 8 KB  K tiles
    __shared__ __align__(16) __bf16 vsm[2][64 * 32];   // 8 KB  V^T tiles
    __shared__ __align__(16) __bf16 pbuf[8][16 * 32];  // 8 KB  per-wave P
    __shared__ float remb[32 * 16];
    for (int i = threadIdx.x; i < 512; i += 256) remb[i] = rel[i];

    const int tid  = threadIdx.x;
    const int lane = tid & 31;
    const int wid  = tid >> 5;
    const int gw   = blockIdx.x * 8 + wid;   // 8192 waves = 4*16*128
    const int b    = gw >> 11;
    const int h    = (gw >> 7) & 15;
    const int qt   = gw & 127;
    const int bh   = b * 16 + h;             // same for all waves in block
    const int hi   = lane >> 4, nl = lane & 15;

    const __bf16* Kbh  = K  + (long)bh * 2048 * 64;
    const __bf16* Vtbh = Vt + (long)bh * 64 * 2048;

    unsigned kofs[2] = { (unsigned)(size_t)&ksm[0][0], (unsigned)(size_t)&ksm[1][0] };
    unsigned vofs[2] = { (unsigned)(size_t)&vsm[0][0], (unsigned)(size_t)&vsm[1][0] };

    const int vrow = tid >> 2, vcol = tid & 3;   // V^T: 64 rows x 4 16B chunks

    auto stage = [&](int kc, int buf) {
        // K tile: contiguous 4KB; thread copies bytes [tid*16, tid*16+16)
        async_b128(kofs[buf] + tid * 16,
                   (unsigned long long)(size_t)((const char*)(Kbh + (long)kc * 32 * 64) + tid * 16));
        // V^T tile: row vrow (hd), 16B sub-chunk vcol; repacked contiguous
        async_b128(vofs[buf] + tid * 16,
                   (unsigned long long)(size_t)(Vtbh + (long)vrow * 2048 + kc * 32 + vcol * 8));
    };

    const __bf16* qbase = Q + ((long)bh * 2048 + qt * 16) * 64;
    v16bf qa0 = load_frag(qbase,      64, lane);
    v16bf qa1 = load_frag(qbase + 32, 64, lane);

    float m_i[8], l_i[8];
    v8f o[4] = {};
#pragma unroll
    for (int r = 0; r < 8; ++r) { m_i[r] = -3.0e38f; l_i[r] = 0.f; }

    stage(0, 0);                               // prologue: 2 async ops/thread
    __syncthreads();                           // remb ready

    for (int kc = 0; kc < 64; ++kc) {          // 32 keys per chunk
        const int cur = kc & 1;
        if (kc + 1 < 64) {
            stage(kc + 1, cur ^ 1);            // overlap copy with compute
            asm volatile("s_wait_asynccnt 0x2" ::: "memory");
        } else {
            asm volatile("s_wait_asynccnt 0x0" ::: "memory");
        }
        __syncthreads();                       // all waves' copies landed

        v8f st[2];
        float cmax[8];
#pragma unroll
        for (int r = 0; r < 8; ++r) cmax[r] = -3.0e38f;

#pragma unroll
        for (int t = 0; t < 2; ++t) {
            const __bf16* kb = &ksm[cur][t * 16 * 64];
            v16bf k0 = load_frag(kb,      64, lane);
            v16bf k1 = load_frag(kb + 32, 64, lane);
            v8f s = {};
            s = wmma_bf16(qa0, k0, s);
            s = wmma_bf16(qa1, k1, s);
            int   j     = kc * 32 + t * 16 + nl;
            float mbias = mask[b * 2048 + j] ? 0.f : -1e9f;
#pragma unroll
            for (int r = 0; r < 8; ++r) {
                int i = qt * 16 + hi * 8 + r;
                s[r] += rel_bias(j - i, h, remb) + mbias;
                cmax[r] = fmaxf(cmax[r], s[r]);
            }
            st[t] = s;
        }

        // online softmax (rows live in 16-lane halves)
#pragma unroll
        for (int r = 0; r < 8; ++r) {
#pragma unroll
            for (int m = 8; m >= 1; m >>= 1)
                cmax[r] = fmaxf(cmax[r], __shfl_xor(cmax[r], m, 32));
            float mnew  = fmaxf(m_i[r], cmax[r]);
            float alpha = __expf(m_i[r] - mnew);
            m_i[r] = mnew;
            l_i[r] *= alpha;
#pragma unroll
            for (int s2 = 0; s2 < 4; ++s2) o[s2][r] *= alpha;
        }

        // P = exp(S - m): C-layout -> A-layout transpose through LDS
        float psum[8];
#pragma unroll
        for (int r = 0; r < 8; ++r) psum[r] = 0.f;
#pragma unroll
        for (int t = 0; t < 2; ++t) {
#pragma unroll
            for (int r = 0; r < 8; ++r) {
                float p = __expf(st[t][r] - m_i[r]);
                psum[r] += p;
                pbuf[wid][(hi * 8 + r) * 32 + t * 16 + nl] = (__bf16)p;
            }
        }
#pragma unroll
        for (int r = 0; r < 8; ++r) {
#pragma unroll
            for (int m = 8; m >= 1; m >>= 1)
                psum[r] += __shfl_xor(psum[r], m, 32);
            l_i[r] += psum[r];
        }

        asm volatile("s_wait_dscnt 0x0" ::: "memory");   // P visible (own wave)

        v16bf pa = load_frag(&pbuf[wid][0], 32, lane);   // P as A(16x32)
#pragma unroll
        for (int s2 = 0; s2 < 4; ++s2) {
            v16bf vf = load_frag(&vsm[cur][s2 * 16 * 32], 32, lane);
            o[s2] = wmma_bf16(pa, vf, o[s2]);
        }
        __syncthreads();   // all reads of buffers done before next overwrite
    }

#pragma unroll
    for (int r = 0; r < 8; ++r) {
        float inv = 1.0f / l_i[r];
        int lq = qt * 16 + hi * 8 + r;
#pragma unroll
        for (int s2 = 0; s2 < 4; ++s2) {
            int d = h * 64 + s2 * 16 + nl;
            ctx[((long)b * 2048 + lq) * 1024 + d] = (__bf16)(o[s2][r] * inv);
        }
    }
}

// ---- launch --------------------------------------------------------------
extern "C" void kernel_launch(void* const* d_in, const int* in_sizes, int n_in,
                              void* d_out, int out_size, void* d_ws, size_t ws_size,
                              hipStream_t stream) {
    const float* hs   = (const float*)d_in[0];
    const int*   mask = (const int*)d_in[1];
    const float* wq   = (const float*)d_in[2];
    const float* wk   = (const float*)d_in[3];
    const float* wv   = (const float*)d_in[4];
    const float* wo   = (const float*)d_in[5];
    const float* rel  = (const float*)d_in[6];
    float* out = (float*)d_out;

    char* ws = (char*)d_ws;                         // 88 MB used
    __bf16* hsb  = (__bf16*)(ws);                   // 16 MB  [B,L,D]
    __bf16* wqb  = (__bf16*)(ws + (16ull << 20));
    __bf16* wkb  = (__bf16*)(ws + (18ull << 20));
    __bf16* wvb  = (__bf16*)(ws + (20ull << 20));
    __bf16* wob  = (__bf16*)(ws + (22ull << 20));
    __bf16* Qb   = (__bf16*)(ws + (24ull << 20));   // [B,H,L,Hd] pre-scaled
    __bf16* Kb   = (__bf16*)(ws + (40ull << 20));   // [B,H,L,Hd]
    __bf16* Vtb  = (__bf16*)(ws + (56ull << 20));   // [B,H,Hd,L]
    __bf16* ctxb = (__bf16*)(ws + (72ull << 20));   // [B,L,D]

    k_cvt<<<8192, 256, 0, stream>>>(hs, hsb, 8192 * 1024);
    k_cvt<<<1024, 256, 0, stream>>>(wq, wqb, 1024 * 1024);
    k_cvt<<<1024, 256, 0, stream>>>(wk, wkb, 1024 * 1024);
    k_cvt<<<1024, 256, 0, stream>>>(wv, wvb, 1024 * 1024);
    k_cvt<<<1024, 256, 0, stream>>>(wo, wob, 1024 * 1024);

    k_gemm_qkv<<<512, 256, 0, stream>>>(hsb, wqb, Qb, 0.125f, 0);
    k_gemm_qkv<<<512, 256, 0, stream>>>(hsb, wkb, Kb, 1.0f,   0);
    k_gemm_qkv<<<512, 256, 0, stream>>>(hsb, wvb, Vtb, 1.0f,  1);

    k_attn<<<1024, 256, 0, stream>>>(Qb, Kb, Vtb, mask, rel, ctxb);

    k_gemm_out<<<512, 256, 0, stream>>>(ctxb, wob, out);
}
